// DglAggregator_40845138985477
// MI455X (gfx1250) — matrix operand order repaired and checked by reading, not compiled, not run on hardware
//
#include <hip/hip_runtime.h>
#include <math.h>

#define DIM    128
#define K2     (2*DIM)      // 256
#define ALPHA  0.2f
#define WAVES  4            // waves per block in stage-2 kernel

typedef __attribute__((ext_vector_type(16))) __bf16      v16bf;
typedef __attribute__((ext_vector_type(8)))  float       v8f;
typedef __attribute__((ext_vector_type(4)))  unsigned int u32x4;

union BF16x16 { u32x4 q[2]; v16bf v; };

// ---- helpers -------------------------------------------------------------
__device__ __forceinline__ unsigned short f2bf(float f) {   // f32 -> bf16 RNE
  unsigned u = __float_as_uint(f);
  unsigned r = u + 0x7FFFu + ((u >> 16) & 1u);
  return (unsigned short)(r >> 16);
}
// order-preserving float<->uint encoding for atomicMax-based segment max
__device__ __forceinline__ unsigned fenc(float f) {
  unsigned u = __float_as_uint(f);
  return (u & 0x80000000u) ? ~u : (u | 0x80000000u);
}
__device__ __forceinline__ float fdec(unsigned u) {
  unsigned v = (u & 0x80000000u) ? (u & 0x7FFFFFFFu) : ~u;
  return __uint_as_float(v);
}

// ---- init: zero ft/denom/out, emax_enc = 0 (== below every finite float) --
__global__ void k_init(float* ftv, float* denom, unsigned* emax, float* out,
                       int n_items, int n_targets) {
  long stride = (long)gridDim.x * blockDim.x;
  long i = (long)blockIdx.x * blockDim.x + threadIdx.x;
  long nft = (long)n_items * DIM, nout = (long)n_targets * DIM;
  for (long t = i; t < nft;  t += stride) ftv[t] = 0.f;
  for (long t = i; t < nout; t += stride) out[t] = 0.f;
  for (long t = i; t < n_items; t += stride) { denom[t] = 0.f; emax[t] = 0u; }
}

// ---- q_w f32[128][256] -> bf16 copy (same row-major layout) ---------------
__global__ void k_qconv(const float* __restrict__ q_w, unsigned short* __restrict__ qb) {
  int i = blockIdx.x * blockDim.x + threadIdx.x;
  if (i < DIM * K2) qb[i] = f2bf(q_w[i]);
}

// ---- stage 1a: e = leakyrelu(sum_d hn[s]*hn[d]*pw[d]); segment max --------
__global__ void k_edge_score(const float* __restrict__ h_n, const float* __restrict__ p_w,
                             const int* __restrict__ src, const int* __restrict__ dst,
                             float* __restrict__ eBuf, unsigned* __restrict__ emax, int E) {
  int wave = threadIdx.x >> 5, lane = threadIdx.x & 31;
  int e = blockIdx.x * 8 + wave;          // blockDim.x == 256 -> 8 waves
  if (e >= E) return;                     // wave-uniform
  int s = src[e], d = dst[e];
  float4 a = ((const float4*)h_n)[s * 32 + lane];
  float4 b = ((const float4*)h_n)[d * 32 + lane];
  float4 w = ((const float4*)p_w)[lane];
  float p = a.x*b.x*w.x + a.y*b.y*w.y + a.z*b.z*w.z + a.w*b.w*w.w;
  #pragma unroll
  for (int off = 16; off; off >>= 1) p += __shfl_xor(p, off, 32);
  if (lane == 0) {
    float v = p > 0.f ? p : ALPHA * p;
    eBuf[e] = v;
    atomicMax(&emax[d], fenc(v));
  }
}

// ---- stage 1b: ee = exp(e - max); denom += ee -----------------------------
__global__ void k_edge_exp(const int* __restrict__ dst, float* __restrict__ eBuf,
                           const unsigned* __restrict__ emax, float* __restrict__ denom, int E) {
  int e = blockIdx.x * blockDim.x + threadIdx.x;
  if (e >= E) return;
  int d = dst[e];
  float ee = __expf(eBuf[e] - fdec(emax[d]));
  eBuf[e] = ee;
  atomicAdd(&denom[d], ee);
}

// ---- stage 1c: ft[dst] += h_n[src] * (ee/denom) ---------------------------
__global__ void k_ft_scatter(const float* __restrict__ h_n,
                             const int* __restrict__ src, const int* __restrict__ dst,
                             const float* __restrict__ eBuf, const float* __restrict__ denom,
                             float* __restrict__ ftv, int E) {
  int wave = threadIdx.x >> 5, lane = threadIdx.x & 31;
  int e = blockIdx.x * 8 + wave;
  if (e >= E) return;                     // wave-uniform
  int s = src[e], d = dst[e];
  float dn = denom[d]; dn = (dn == 0.f) ? 1.f : dn;
  float a = eBuf[e] / dn;
  float4 hv = ((const float4*)h_n)[s * 32 + lane];
  float* o = ftv + (long)d * DIM + lane * 4;
  atomicAdd(o + 0, hv.x * a); atomicAdd(o + 1, hv.y * a);
  atomicAdd(o + 2, hv.z * a); atomicAdd(o + 3, hv.w * a);
}

// ---- stage 2: WMMA GEMM [16x256]x[256x128] per wave + epilogue ------------
__global__ void __launch_bounds__(WAVES * 32)
k_stage2(const float* __restrict__ ftv, const float* __restrict__ h_p,
         const float* __restrict__ h_t, const unsigned short* __restrict__ qb,
         const int* __restrict__ src_a, const int* __restrict__ dst_a,
         float* __restrict__ out, int E) {
  __shared__ unsigned short sA[WAVES][16][K2];   // bf16 A tiles: 32 KB
  __shared__ int   sSrc[WAVES][16];
  __shared__ int   sDst[WAVES][16];
  __shared__ float sCoef[WAVES][16];

  const int wave = threadIdx.x >> 5;
  const int lane = threadIdx.x & 31;
  const int hi   = lane >> 4;        // half-wave select (CDNA5 WMMA layouts)
  const int nn   = lane & 15;
  const int ebase = (blockIdx.x * WAVES + wave) * 16;

  __builtin_prefetch(qb + (threadIdx.x << 5), 0, 0);   // global_prefetch_b8 on q table

  // stage A tile rows into LDS as bf16: cols [0,128) = ft[src], [128,256) = h_p[e]
  for (int m = 0; m < 16; ++m) {
    int e = ebase + m;
    int s = 0, d = -1;
    if (e < E) { s = src_a[e]; d = dst_a[e]; }         // wave-uniform branch
    if (lane == 0) { sSrc[wave][m] = s; sDst[wave][m] = d; }
    float4 f1 = make_float4(0.f, 0.f, 0.f, 0.f), f2 = f1;
    if (e < E) {
      f1 = ((const float4*)ftv)[s * 32 + lane];
      f2 = ((const float4*)h_p)[(long)e * 32 + lane];
    }
    unsigned short* row = &sA[wave][m][0];
    int c = lane * 4;
    row[c + 0] = f2bf(f1.x); row[c + 1] = f2bf(f1.y);
    row[c + 2] = f2bf(f1.z); row[c + 3] = f2bf(f1.w);
    row[DIM + c + 0] = f2bf(f2.x); row[DIM + c + 1] = f2bf(f2.y);
    row[DIM + c + 2] = f2bf(f2.z); row[DIM + c + 3] = f2bf(f2.w);
  }
  __syncthreads();

  // GEMM: acc[m=16][n=128] f32 = A(16x256,bf16) * qb^T ; 8 K-chunks x 8 N-tiles
  v8f acc[8];
  #pragma unroll
  for (int nt = 0; nt < 8; ++nt) acc[nt] = (v8f){0.f,0.f,0.f,0.f,0.f,0.f,0.f,0.f};

  #pragma unroll
  for (int kc = 0; kc < 8; ++kc) {
    // A layout (16-bit 16x32): lane half 0 -> K runs [0..7],[16..23]; half 1 -> +8
    const unsigned short* arow = &sA[wave][nn][0];     // row m = lane & 15
    BF16x16 afrag;
    afrag.q[0] = *(const u32x4*)(arow + kc * 32 + hi * 8);
    afrag.q[1] = *(const u32x4*)(arow + kc * 32 + 16 + hi * 8);
    #pragma unroll
    for (int nt = 0; nt < 8; ++nt) {
      // B layout (32x16): lane n = lane&15, K runs [0..7],[16..23] for half 0; half 1 -> +8
      const unsigned short* brow = qb + (nt * 16 + nn) * K2 + kc * 32 + hi * 8;
      BF16x16 bfrag;
      bfrag.q[0] = ((const u32x4*)brow)[0];
      bfrag.q[1] = ((const u32x4*)(brow + 16))[0];
      acc[nt] = __builtin_amdgcn_wmma_f32_16x16x32_bf16(
          false, afrag.v, false, bfrag.v, (short)0, acc[nt], false, false);
    }
  }

  // coef[m] = sum_n tanh(e2[m,n]) * h_t[dst[m],n] ; C layout: elem r -> row r+8*hi
  #pragma unroll
  for (int r = 0; r < 8; ++r) {
    int m = r + 8 * hi;
    int d = sDst[wave][m];
    int dS = d < 0 ? 0 : d;
    const float* ht = h_t + (long)dS * DIM + nn;
    float p = 0.f;
    #pragma unroll
    for (int nt = 0; nt < 8; ++nt) p += tanhf(acc[nt][r]) * ht[nt * 16];
    p += __shfl_xor(p, 8, 16);
    p += __shfl_xor(p, 4, 16);
    p += __shfl_xor(p, 2, 16);
    p += __shfl_xor(p, 1, 16);
    if (nn == 0) sCoef[wave][m] = p;
  }
  __syncthreads();

  // out[dst[m]] += ft[src[m]] * coef[m]
  for (int m = 0; m < 16; ++m) {
    int e = ebase + m;
    if (e >= E) continue;                               // wave-uniform
    int d = sDst[wave][m];
    if (d < 0) continue;
    float coef = sCoef[wave][m];
    int s = sSrc[wave][m];
    float4 f = ((const float4*)ftv)[s * 32 + lane];
    float* o = out + (long)d * DIM + lane * 4;
    atomicAdd(o + 0, f.x * coef); atomicAdd(o + 1, f.y * coef);
    atomicAdd(o + 2, f.z * coef); atomicAdd(o + 3, f.w * coef);
  }
}

// ---- host launcher --------------------------------------------------------
extern "C" void kernel_launch(void* const* d_in, const int* in_sizes, int n_in,
                              void* d_out, int out_size, void* d_ws, size_t ws_size,
                              hipStream_t stream) {
  const float* h_n = (const float*)d_in[0];
  const float* h_p = (const float*)d_in[1];
  const float* h_t = (const float*)d_in[2];
  const float* p_w = (const float*)d_in[3];
  const float* q_w = (const float*)d_in[4];
  const int* src_i = (const int*)d_in[5];
  const int* dst_i = (const int*)d_in[6];
  const int* src_a = (const int*)d_in[7];
  const int* dst_a = (const int*)d_in[8];
  float* out = (float*)d_out;

  const int n_items   = in_sizes[0] / DIM;
  const int n_targets = in_sizes[2] / DIM;
  const int E_int     = in_sizes[5];
  const int E_agg     = in_sizes[7];

  // carve workspace (~29.3 MB total)
  auto au = [](size_t x) { return (x + 255) & ~(size_t)255; };
  char* w = (char*)d_ws;
  float*          eBuf  = (float*)w;          w += au((size_t)E_int * 4);
  unsigned*       emax  = (unsigned*)w;       w += au((size_t)n_items * 4);
  float*          denom = (float*)w;          w += au((size_t)n_items * 4);
  float*          ftv   = (float*)w;          w += au((size_t)n_items * DIM * 4);
  unsigned short* qb    = (unsigned short*)w; w += au((size_t)DIM * K2 * 2);
  (void)ws_size; (void)n_in; (void)out_size;

  k_init<<<2048, 256, 0, stream>>>(ftv, denom, emax, out, n_items, n_targets);
  k_qconv<<<(DIM * K2 + 255) / 256, 256, 0, stream>>>(q_w, qb);
  k_edge_score<<<(E_int + 7) / 8, 256, 0, stream>>>(h_n, p_w, src_i, dst_i, eBuf, emax, E_int);
  k_edge_exp<<<(E_int + 255) / 256, 256, 0, stream>>>(dst_i, eBuf, emax, denom, E_int);
  k_ft_scatter<<<(E_int + 7) / 8, 256, 0, stream>>>(h_n, src_i, dst_i, eBuf, denom, ftv, E_int);
  k_stage2<<<(E_agg + WAVES * 16 - 1) / (WAVES * 16), WAVES * 32, 0, stream>>>(
      ftv, h_p, h_t, qb, src_a, dst_a, out, E_agg);
}
